// CrossAttentionModel_75118978007258
// MI455X (gfx1250) — compile-verified, matrix operand order checked by reading, not run on hardware
//
#include <hip/hip_runtime.h>
#include <hip/hip_bf16.h>
#include <math.h>
#include <stdint.h>

// ---------------- problem constants ----------------
constexpr int NN    = 65536;    // nodes
constexpr int EE    = 524288;   // edges
constexpr int GG    = 1024;     // graphs
constexpr int RR    = 512;      // residues
constexpr int HH    = 128;      // hidden
constexpr int NHEAD = 4;
constexpr int HDIM  = 32;
constexpr float EPSV = 1e-5f;

// ---------------- WMMA types (CDNA5, wave32) ----------------
typedef __attribute__((ext_vector_type(16))) __bf16 v16bf;
typedef __attribute__((ext_vector_type(8)))  float  v8f;

#define WMMA_BF16(a, b, c) \
  __builtin_amdgcn_wmma_f32_16x16x32_bf16(false, (a), false, (b), (short)0, (c), false, false)

// A fragment: 16x32 bf16.  Lane L: m = L&15, h = L>>4.
//   elements 0..7  -> K = 8*h + e
//   elements 8..15 -> K = 16 + 8*h + (e-8)
__device__ __forceinline__ v16bf frag_a_rows(const float* base, int ld, int k0) {
  const int lane = threadIdx.x & 31;
  const int m = lane & 15, h = (lane >> 4) & 1;
  const float* p = base + (size_t)m * ld + k0 + 8 * h;
  v16bf a;
#pragma unroll
  for (int e = 0; e < 8; ++e) a[e] = (__bf16)p[e];
#pragma unroll
  for (int e = 0; e < 8; ++e) a[8 + e] = (__bf16)p[16 + e];
  return a;
}

// B fragment: 32x16 bf16 from row-major B[k][n] (global, ld = row stride).
__device__ __forceinline__ v16bf frag_b_nn(const float* B, int ld, int k0, int col0) {
  const int lane = threadIdx.x & 31;
  const int n = lane & 15, h = (lane >> 4) & 1;
  const float* p = B + (size_t)(k0 + 8 * h) * ld + col0 + n;
  v16bf b;
#pragma unroll
  for (int e = 0; e < 8; ++e) b[e] = (__bf16)p[(size_t)e * ld];
#pragma unroll
  for (int e = 0; e < 8; ++e) b[8 + e] = (__bf16)p[(size_t)(16 + e) * ld];
  return b;
}

// B fragment where effective B[k][n] = Bt[(col0+n)*ld + k]  (global transposed).
__device__ __forceinline__ v16bf frag_b_t(const float* Bt, int ld, int k0, int col0) {
  const int lane = threadIdx.x & 31;
  const int n = lane & 15, h = (lane >> 4) & 1;
  const float* p = Bt + (size_t)(col0 + n) * ld + k0 + 8 * h;
  v16bf b;
#pragma unroll
  for (int e = 0; e < 8; ++e) b[e] = (__bf16)p[e];
#pragma unroll
  for (int e = 0; e < 8; ++e) b[8 + e] = (__bf16)p[16 + e];
  return b;
}

// B fragments from an LDS-resident 128x128 f32 block.
__device__ __forceinline__ v16bf frag_b_lds_nn(const float (*lB)[HH], int k0, int col0) {
  const int lane = threadIdx.x & 31;
  const int n = lane & 15, h = (lane >> 4) & 1;
  v16bf b;
#pragma unroll
  for (int e = 0; e < 8; ++e) b[e] = (__bf16)lB[k0 + 8 * h + e][col0 + n];
#pragma unroll
  for (int e = 0; e < 8; ++e) b[8 + e] = (__bf16)lB[k0 + 16 + 8 * h + e][col0 + n];
  return b;
}
__device__ __forceinline__ v16bf frag_b_lds_t(const float (*lB)[HH], int k0, int col0) {
  const int lane = threadIdx.x & 31;
  const int n = lane & 15, h = (lane >> 4) & 1;
  const int kk = k0 + 8 * h;
  v16bf b;
#pragma unroll
  for (int e = 0; e < 8; ++e) b[e] = (__bf16)lB[col0 + n][kk + e];
#pragma unroll
  for (int e = 0; e < 8; ++e) b[8 + e] = (__bf16)lB[col0 + n][kk + 16 + e];
  return b;
}

// C/D 16x16 f32: lane L holds column n = L&15, rows (L>>4)*8 + v.
__device__ __forceinline__ void store_c(float* Cbase, int ld, v8f acc) {
  const int lane = threadIdx.x & 31;
  const int n = lane & 15, mh = (lane >> 4) & 1;
#pragma unroll
  for (int v = 0; v < 8; ++v) Cbase[(size_t)(mh * 8 + v) * ld + n] = acc[v];
}

// ---------------- degree / norm ----------------
__global__ void deg_init(float* deg) {
  int i = blockIdx.x * blockDim.x + threadIdx.x;
  if (i < NN) deg[i] = 1.0f;
}
__global__ void deg_accum(const int* __restrict__ dst, float* deg) {
  int e = blockIdx.x * blockDim.x + threadIdx.x;
  if (e < EE) atomicAdd(&deg[dst[e]], 1.0f);
}
__global__ void deg_finalize(float* deg) {
  int i = blockIdx.x * blockDim.x + threadIdx.x;
  if (i < NN) deg[i] = rsqrtf(deg[i]);   // deg -> dinv in place
}

// ---------------- layer-0 GEMM (K=7, scalar) ----------------
__global__ void gemm_in(const float* __restrict__ x, const float* __restrict__ W0,
                        float* __restrict__ out) {
  size_t i = (size_t)blockIdx.x * blockDim.x + threadIdx.x;  // N*H
  int n = (int)(i >> 7), c = (int)(i & 127);
  float acc = 0.f;
#pragma unroll
  for (int d = 0; d < 7; ++d) acc += x[(size_t)n * 7 + d] * W0[d * HH + c];
  out[i] = acc;
}

// ---------------- LDS-staged WMMA GEMM --------------------------------------
// C[M,128] = A[M,128] @ Beff[128,128] (+bias).  TRANSB: Beff[k][n] = B[n*128+k].
// Block: 256 threads = 8 waves, 128x128 output slab, K = 128.
// The 64 KB f32 weight block is staged global->LDS with CDNA5 async copies
// (ASYNCcnt), then each wave reuses one A fragment across 8 column tiles.
template <bool TRANSB, bool BIAS>
__global__ void wmma_gemm_lds(const float* __restrict__ A, const float* __restrict__ B,
                              const float* __restrict__ bias, float* __restrict__ C) {
  __shared__ float lB[HH][HH];                       // 64 KB of the 320 KB WGP LDS

  // ---- async stage: 4096 x b128 lane-copies, tracked by ASYNCcnt ----
  {
    const int tid = threadIdx.y * 32 + threadIdx.x;  // 0..255
    const unsigned lbase = (unsigned)(uintptr_t)(void*)&lB[0][0];
#pragma unroll
    for (int it = 0; it < 16; ++it) {
      const int idx = tid + it * 256;                // 16-byte chunk index
      const float* gp = B + (size_t)idx * 4;
      const unsigned loff = lbase + (unsigned)idx * 16u;
      asm volatile("global_load_async_to_lds_b128 %0, %1, off"
                   :: "v"(loff), "v"(gp) : "memory");
    }
    asm volatile("s_wait_asynccnt 0" ::: "memory");
    __syncthreads();
  }

  const int row0 = blockIdx.x * 128 + threadIdx.y * 16;
  v8f acc[8] = {};
#pragma unroll
  for (int k0 = 0; k0 < HH; k0 += 32) {
    const v16bf a = frag_a_rows(A + (size_t)row0 * HH, HH, k0);   // global, streamed
#pragma unroll
    for (int t = 0; t < 8; ++t) {
      v16bf b = TRANSB ? frag_b_lds_t(lB, k0, t * 16)
                       : frag_b_lds_nn(lB, k0, t * 16);
      acc[t] = WMMA_BF16(a, b, acc[t]);
    }
  }

  const int lane = threadIdx.x & 31;
  const int n = lane & 15, mh = (lane >> 4) & 1;
#pragma unroll
  for (int t = 0; t < 8; ++t) {
    float bv = 0.f;
    if constexpr (BIAS) bv = bias[t * 16 + n];
    float* cb = C + (size_t)(row0 + mh * 8) * HH + t * 16 + n;
#pragma unroll
    for (int v = 0; v < 8; ++v) cb[(size_t)v * HH] = acc[t][v] + bv;
  }
}

// ---------------- GCN aggregation ----------------
__global__ void scale_self(const float* __restrict__ m, const float* __restrict__ dinv,
                           float* __restrict__ agg) {
  size_t i = (size_t)blockIdx.x * blockDim.x + threadIdx.x;
  int n = (int)(i >> 7);
  float d = dinv[n];
  agg[i] = m[i] * d * d;
}

__global__ void edge_scatter(const float* __restrict__ m, const int* __restrict__ src,
                             const int* __restrict__ dst, const float* __restrict__ dinv,
                             float* __restrict__ agg) {
  const int e = blockIdx.x;
  const int c = threadIdx.x;
  const int s = src[e], d = dst[e];
  const float w = dinv[s] * dinv[d];
  atomicAdd(&agg[(size_t)d * HH + c], m[(size_t)s * HH + c] * w);
}

__global__ void bias_bn_relu(const float* __restrict__ agg, const float* __restrict__ b,
                             const float* __restrict__ gamma, const float* __restrict__ beta,
                             const float* __restrict__ mean, const float* __restrict__ var,
                             float* __restrict__ out) {
  size_t i = (size_t)blockIdx.x * blockDim.x + threadIdx.x;
  int c = (int)(i & 127);
  float y = agg[i] + b[c];
  y = gamma[c] * (y - mean[c]) * rsqrtf(var[c] + EPSV) + beta[c];
  out[i] = fmaxf(y, 0.f);
}

// ---------------- residue projection (K=25, scalar) ----------------
__global__ void res_proj(const float* __restrict__ prf, const float* __restrict__ rW,
                         const float* __restrict__ rb, float* __restrict__ res) {
  size_t i = (size_t)blockIdx.x * blockDim.x + threadIdx.x;  // R*H
  int r = (int)(i >> 7), c = (int)(i & 127);
  float acc = rb[c];
#pragma unroll
  for (int j = 0; j < 25; ++j) acc += prf[(size_t)r * 25 + j] * rW[j * HH + c];
  res[i] = acc;
}

// ---------------- fused cross-attention (one wave, 16 nodes x 1 head) --------
// scores (32 WMMAs, K=32) -> LDS -> softmax over R=512 -> attn@V (32 WMMAs).
__global__ void attention(const float* __restrict__ q, const float* __restrict__ kbuf,
                          const float* __restrict__ vbuf, float* __restrict__ o) {
  __shared__ float S[16][RR];            // 32 KB
  const int node0 = blockIdx.x * 16;
  const int head  = blockIdx.y;
  const int lane  = threadIdx.x;
  const float scale = 0.17677669529663687f;  // 1/sqrt(32)

  const v16bf aQ = frag_a_rows(q + (size_t)node0 * HH, HH, head * HDIM);

#pragma unroll 4
  for (int rt = 0; rt < RR / 16; ++rt) {
    v16bf bK = frag_b_t(kbuf, HH, head * HDIM, rt * 16);
    v8f acc = {};
    acc = WMMA_BF16(aQ, bK, acc);
    const int n = lane & 15, mh = (lane >> 4) & 1;
#pragma unroll
    for (int v = 0; v < 8; ++v) S[mh * 8 + v][rt * 16 + n] = acc[v] * scale;
  }
  __syncthreads();

  if (lane < 16) {                        // one lane per query row
    float mx = -3.4e38f;
    for (int r = 0; r < RR; ++r) mx = fmaxf(mx, S[lane][r]);
    float sum = 0.f;
    for (int r = 0; r < RR; ++r) { float p = __expf(S[lane][r] - mx); S[lane][r] = p; sum += p; }
    const float inv = 1.f / sum;
    for (int r = 0; r < RR; ++r) S[lane][r] *= inv;
  }
  __syncthreads();

  v8f acc0 = {}, acc1 = {};
#pragma unroll 4
  for (int rb = 0; rb < RR / 32; ++rb) {
    v16bf aP  = frag_a_rows(&S[0][0], RR, rb * 32);           // probs from LDS
    v16bf bV0 = frag_b_nn(vbuf, HH, rb * 32, head * HDIM);
    v16bf bV1 = frag_b_nn(vbuf, HH, rb * 32, head * HDIM + 16);
    acc0 = WMMA_BF16(aP, bV0, acc0);
    acc1 = WMMA_BF16(aP, bV1, acc1);
  }
  store_c(o + (size_t)node0 * HH + head * HDIM,      HH, acc0);
  store_c(o + (size_t)node0 * HH + head * HDIM + 16, HH, acc1);
}

// ---------------- per-graph mean pool (64 sorted nodes / graph) --------------
__global__ void pool_mean(const float* __restrict__ a, float* __restrict__ pooled) {
  const int g = blockIdx.x, c = threadIdx.x;
  const float* p = a + (size_t)g * 64 * HH + c;
  float s = 0.f;
#pragma unroll 8
  for (int i = 0; i < 64; ++i) s += p[(size_t)i * HH];
  pooled[(size_t)g * HH + c] = s * (1.f / 64.f);
}

// ---------------- classifier 128 -> 64 -> 1 ----------------
__global__ void classify(const float* __restrict__ pooled, const float* __restrict__ W1c,
                         const float* __restrict__ b1c, const float* __restrict__ W2c,
                         const float* __restrict__ b2c, float* __restrict__ out) {
  __shared__ float red[64];
  const int g = blockIdx.x, j = threadIdx.x;
  float acc = b1c[j];
#pragma unroll 8
  for (int c = 0; c < HH; ++c) acc += pooled[(size_t)g * HH + c] * W1c[c * 64 + j];
  red[j] = fmaxf(acc, 0.f) * W2c[j];
  __syncthreads();
  for (int s = 32; s > 0; s >>= 1) {
    if (j < s) red[j] += red[j + s];
    __syncthreads();
  }
  if (j == 0) out[g] = red[0] + b2c[0];
}

// ---------------- launch ----------------
extern "C" void kernel_launch(void* const* d_in, const int* in_sizes, int n_in,
                              void* d_out, int out_size, void* d_ws, size_t ws_size,
                              hipStream_t stream) {
  const float* x     = (const float*)d_in[0];
  const int*   eidx  = (const int*)d_in[1];
  const float* prf   = (const float*)d_in[3];
  const float* W0    = (const float*)d_in[4];
  const float* b0    = (const float*)d_in[5];
  const float* W1    = (const float*)d_in[6];
  const float* b1    = (const float*)d_in[7];
  const float* W2    = (const float*)d_in[8];
  const float* b2    = (const float*)d_in[9];
  const float* bng   = (const float*)d_in[10];
  const float* bnb   = (const float*)d_in[11];
  const float* bnm   = (const float*)d_in[12];
  const float* bnv   = (const float*)d_in[13];
  const float* resW  = (const float*)d_in[14];
  const float* resb  = (const float*)d_in[15];
  const float* inW   = (const float*)d_in[16];
  const float* inb   = (const float*)d_in[17];
  const float* outW  = (const float*)d_in[18];
  const float* outb  = (const float*)d_in[19];
  const float* cW1   = (const float*)d_in[20];
  const float* cb1   = (const float*)d_in[21];
  const float* cW2   = (const float*)d_in[22];
  const float* cb2   = (const float*)d_in[23];
  float* out = (float*)d_out;

  const int* src = eidx;
  const int* dst = eidx + EE;

  const size_t NHf = (size_t)NN * HH;
  float* ws   = (float*)d_ws;
  float* B0   = ws;                 // h
  float* B1   = ws + NHf;           // m / q / attn_out
  float* B2   = ws + 2 * NHf;       // agg / o
  float* dinv = ws + 3 * NHf;       // N
  float* res  = dinv + NN;          // R*H
  float* kb   = res + (size_t)RR * HH;
  float* vb   = kb + (size_t)RR * HH;
  float* pooled = vb + (size_t)RR * HH;   // G*H

  const dim3 gemmBlk(32, 8);        // 8 waves, 128x128 slab
  const int  gemmGridN = NN / 128;
  const int  gemmGridR = RR / 128;
  const int  NHthr = (int)(NHf / 256);

  // degree -> dinv
  deg_init<<<NN / 256, 256, 0, stream>>>(dinv);
  deg_accum<<<EE / 256, 256, 0, stream>>>(dst, dinv);
  deg_finalize<<<NN / 256, 256, 0, stream>>>(dinv);

  // layer 0 (K=7 scalar GEMM)
  gemm_in<<<NHthr, 256, 0, stream>>>(x, W0, B1);
  scale_self<<<NHthr, 256, 0, stream>>>(B1, dinv, B2);
  edge_scatter<<<EE, HH, 0, stream>>>(B1, src, dst, dinv, B2);
  bias_bn_relu<<<NHthr, 256, 0, stream>>>(B2, b0, bng, bnb, bnm, bnv, B0);

  // layer 1 (WMMA, async-staged weights)
  wmma_gemm_lds<false, false><<<gemmGridN, gemmBlk, 0, stream>>>(B0, W1, nullptr, B1);
  scale_self<<<NHthr, 256, 0, stream>>>(B1, dinv, B2);
  edge_scatter<<<EE, HH, 0, stream>>>(B1, src, dst, dinv, B2);
  bias_bn_relu<<<NHthr, 256, 0, stream>>>(B2, b1, bng + HH, bnb + HH, bnm + HH, bnv + HH, B0);

  // layer 2 (WMMA)
  wmma_gemm_lds<false, false><<<gemmGridN, gemmBlk, 0, stream>>>(B0, W2, nullptr, B1);
  scale_self<<<NHthr, 256, 0, stream>>>(B1, dinv, B2);
  edge_scatter<<<EE, HH, 0, stream>>>(B1, src, dst, dinv, B2);
  bias_bn_relu<<<NHthr, 256, 0, stream>>>(B2, b2, bng + 2 * HH, bnb + 2 * HH,
                                          bnm + 2 * HH, bnv + 2 * HH, B0);

  // residues: projection, then K/V via WMMA (in_W rows are output features)
  res_proj<<<(RR * HH) / 256, 256, 0, stream>>>(prf, resW, resb, res);
  wmma_gemm_lds<true, true><<<gemmGridR, gemmBlk, 0, stream>>>(res, inW + HH * HH,     inb + HH,     kb);
  wmma_gemm_lds<true, true><<<gemmGridR, gemmBlk, 0, stream>>>(res, inW + 2 * HH * HH, inb + 2 * HH, vb);

  // Q projection (WMMA), fused attention, out-projection (WMMA)
  wmma_gemm_lds<true, true><<<gemmGridN, gemmBlk, 0, stream>>>(B0, inW, inb, B1);   // q -> B1
  attention<<<dim3(NN / 16, NHEAD), 32, 0, stream>>>(B1, kb, vb, B2);               // o -> B2
  wmma_gemm_lds<true, true><<<gemmGridN, gemmBlk, 0, stream>>>(B2, outW, outb, B1); // attn_out -> B1

  // pool + classifier
  pool_mean<<<GG, HH, 0, stream>>>(B1, pooled);
  classify<<<GG, 64, 0, stream>>>(pooled, cW1, cb1, cW2, cb2, out);

  (void)in_sizes; (void)n_in; (void)out_size; (void)ws_size;
}
